// AutoregressiveFlow_78065325572313
// MI455X (gfx1250) — matrix operand order repaired and checked by reading, not compiled, not run on hardware
//
#include <hip/hip_runtime.h>
#include <hip/hip_bf16.h>
#include <math.h>

typedef _Float16 v16h __attribute__((ext_vector_type(16)));
typedef float    v8f  __attribute__((ext_vector_type(8)));

#define NB      4096      // batch
#define LSEQ    256       // sequence length
#define TSTEPS  255       // L-1
#define HDIM    256
#define MMIX    8
#define OUT3    24
#define ROWS    32        // batch rows per block
#define TWO_PI_F     6.2831853071795864769f
#define INV_TWO_PI_F 0.15915494309189533577f
#define EPS_F        1e-06f

// Pre-swizzled f16 weight buffers in workspace:
//   W1f: [t][nt(16)][kt(8)][lane(32)][e(16)] halves  -> 33,423,360 B
//   W2f: [t][nt2(2)][kt(8)][lane(32)][e(16)] halves  ->  4,177,920 B (24 cols zero-padded to 32)
#define W1F_HALVES ((size_t)TSTEPS * 16 * 8 * 512)
#define W2F_HALVES ((size_t)TSTEPS * 2 * 8 * 512)
#define WS_NEEDED  ((W1F_HALVES + W2F_HALVES) * 2)

__device__ __forceinline__ float softplus_f(float x) {
    return (x > 20.0f) ? x : log1pf(expf(x));
}

// A-fragment scatter (16-bit A, 16x32, wave32; contiguous k order):
//   lane = m + 16*half, half = kk>>4, e = kk&15
__device__ __forceinline__ int afrag_off(int rt, int kt, int m, int kk) {
    const int half = (kk >> 4) & 1;
    const int e    = kk & 15;
    return (rt * 8 + kt) * 512 + (m + 16 * half) * 16 + e;
}

// ---------------- Pre-pass: W1 f32 -> f16 B-fragment order ----------------
__global__ __launch_bounds__(256)
void preswizzle_w1(const float* __restrict__ W1, _Float16* __restrict__ W1f) {
    const int gid  = blockIdx.x * 256 + threadIdx.x;   // (t,nt,kt,lane)
    const int lane = gid & 31;
    const int kt   = (gid >> 5) & 7;
    const int nt   = (gid >> 8) & 15;
    const int t    = gid >> 12;
    if (t >= TSTEPS) return;
    const int n     = nt * 16 + (lane & 15);
    const int khalf = lane >> 4;
    v16h frag;
    #pragma unroll
    for (int e = 0; e < 16; ++e) {
        const int k = kt * 32 + e + 16 * khalf;
        frag[e] = (_Float16)W1[((size_t)t * HDIM + k) * HDIM + n];
    }
    *reinterpret_cast<v16h*>(W1f + (size_t)gid * 16) = frag;
}

__global__ __launch_bounds__(256)
void preswizzle_w2(const float* __restrict__ W2, _Float16* __restrict__ W2f) {
    const int gid  = blockIdx.x * 256 + threadIdx.x;   // (t,nt2,kt,lane)
    const int lane = gid & 31;
    const int kt   = (gid >> 5) & 7;
    const int nt2  = (gid >> 8) & 1;
    const int t    = gid >> 9;
    if (t >= TSTEPS) return;
    const int n2    = nt2 * 16 + (lane & 15);
    const bool ok   = (n2 < OUT3);
    const int khalf = lane >> 4;
    v16h frag;
    #pragma unroll
    for (int e = 0; e < 16; ++e) {
        const int k = kt * 32 + e + 16 * khalf;
        frag[e] = ok ? (_Float16)W2[((size_t)t * HDIM + k) * OUT3 + n2] : (_Float16)0.0f;
    }
    *reinterpret_cast<v16h*>(W2f + (size_t)gid * 16) = frag;
}

// ---------------- Main autoregressive flow kernel ----------------
template <bool PRE>
__global__ __launch_bounds__(256)
void ar_flow_kernel(const float* __restrict__ phi,
                    const float* __restrict__ W0,
                    const float* __restrict__ b0,
                    const float* __restrict__ W1,
                    const float* __restrict__ b1,
                    const float* __restrict__ W2,
                    const float* __restrict__ b2,
                    const _Float16* __restrict__ W1f,
                    const _Float16* __restrict__ W2f,
                    float* __restrict__ outTheta,
                    float* __restrict__ outLdj) {
    __shared__ __attribute__((aligned(32))) _Float16 s_h0[2 * 8 * 512]; // A-frags of h0 (32x256)
    __shared__ __attribute__((aligned(32))) _Float16 s_h1[2 * 8 * 512]; // A-frags of h1 (32x256)
    __shared__ float s_p[ROWS * 32];       // head outputs, padded 24->32
    __shared__ float s_thp[ROWS];          // theta_prev per row
    __shared__ float s_th0[ROWS];          // theta_0 per row

    const int tid     = threadIdx.x;
    const int lane    = tid & 31;
    const int wave    = tid >> 5;          // 0..7
    const int nlane   = lane & 15;
    const int khalf   = lane >> 4;
    const int rowBase = blockIdx.x * ROWS;

    if (tid < ROWS) {
        const float th0 = phi[(size_t)(rowBase + tid) * LSEQ];
        s_th0[tid] = th0;
        s_thp[tid] = th0;
        outTheta[(size_t)(rowBase + tid) * LSEQ] = th0;
    }
    __syncthreads();

    float ldjAcc = 0.0f;
    const int rA = tid & 31;               // row served in phase A
    const int rM = tid >> 3;               // row served in mixture phase
    const int jM = tid & 7;                // mixture component served

    for (int t = 0; t < TSTEPS; ++t) {
        const float* W0row = W0 + (size_t)t * HDIM;
        const float* b0row = b0 + (size_t)t * HDIM;
        const float* W1t   = W1 + (size_t)t * HDIM * HDIM;
        const float* b1row = b1 + (size_t)t * HDIM;
        const float* W2t   = W2 + (size_t)t * HDIM * OUT3;
        const float* b2row = b2 + (size_t)t * OUT3;

        // ---- Phase A: h0 = tanh(ctx * W0 + b0) -> LDS A-fragment layout ----
        {
            const float ctx = s_thp[rA] - s_th0[rA];
            const int rt = rA >> 4, m = rA & 15;
            #pragma unroll 4
            for (int i = 0; i < 32; ++i) {
                const int k = wave + 8 * i;
                const float v = tanhf(ctx * W0row[k] + b0row[k]);
                s_h0[afrag_off(rt, k >> 5, m, k & 31)] = (_Float16)v;
            }
        }
        __syncthreads();

        // ---- GEMM1: h1 = tanh(h0 @ W1 + b1); wave owns N-tiles {2w, 2w+1} ----
        {
            const int nt0 = 2 * wave, nt1 = 2 * wave + 1;
            v8f acc[2][2];   // [ntSel][rt]
            #pragma unroll
            for (int s = 0; s < 2; ++s) {
                const float bb = b1row[(2 * wave + s) * 16 + nlane];
                #pragma unroll
                for (int rt = 0; rt < 2; ++rt)
                    #pragma unroll
                    for (int v = 0; v < 8; ++v) acc[s][rt][v] = bb;
            }
            for (int kt = 0; kt < 8; ++kt) {
                const v16h af0 = *reinterpret_cast<const v16h*>(&s_h0[(0 * 8 + kt) * 512 + lane * 16]);
                const v16h af1 = *reinterpret_cast<const v16h*>(&s_h0[(1 * 8 + kt) * 512 + lane * 16]);
                v16h bf0, bf1;
                if (PRE) {
                    bf0 = *reinterpret_cast<const v16h*>(
                        W1f + (((size_t)t * 16 + nt0) * 8 + kt) * 512 + lane * 16);
                    bf1 = *reinterpret_cast<const v16h*>(
                        W1f + (((size_t)t * 16 + nt1) * 8 + kt) * 512 + lane * 16);
                } else {
                    const float* c0 = W1t + nt0 * 16 + nlane;
                    const float* c1 = W1t + nt1 * 16 + nlane;
                    #pragma unroll
                    for (int e = 0; e < 16; ++e) {
                        const int k = kt * 32 + e + 16 * khalf;
                        bf0[e] = (_Float16)c0[(size_t)k * HDIM];
                        bf1[e] = (_Float16)c1[(size_t)k * HDIM];
                    }
                }
                acc[0][0] = __builtin_amdgcn_wmma_f32_16x16x32_f16(false, af0, false, bf0, (short)0, acc[0][0], false, false);
                acc[0][1] = __builtin_amdgcn_wmma_f32_16x16x32_f16(false, af1, false, bf0, (short)0, acc[0][1], false, false);
                acc[1][0] = __builtin_amdgcn_wmma_f32_16x16x32_f16(false, af0, false, bf1, (short)0, acc[1][0], false, false);
                acc[1][1] = __builtin_amdgcn_wmma_f32_16x16x32_f16(false, af1, false, bf1, (short)0, acc[1][1], false, false);
            }
            // tanh + scatter h1 into A-fragment layout for GEMM2
            #pragma unroll
            for (int s = 0; s < 2; ++s) {
                const int nt  = 2 * wave + s;
                const int kk2 = (nt & 1) * 16 + nlane;
                const int kt2 = nt >> 1;
                #pragma unroll
                for (int rt = 0; rt < 2; ++rt)
                    #pragma unroll
                    for (int v = 0; v < 8; ++v) {
                        const int rl = rt * 16 + v + 8 * khalf;
                        s_h1[afrag_off(rl >> 4, kt2, rl & 15, kk2)] =
                            (_Float16)tanhf(acc[s][rt][v]);
                    }
            }
        }
        __syncthreads();

        // ---- GEMM2: p = h1 @ W2 + b2 (24 cols padded to 32); waves 0..3 ----
        if (wave < 4) {
            const int rt  = wave & 1;
            const int nt2 = wave >> 1;
            const int n2  = nt2 * 16 + nlane;
            const bool ok = (n2 < OUT3);
            v8f acc2;
            {
                const float bb = ok ? b2row[n2] : 0.0f;
                #pragma unroll
                for (int v = 0; v < 8; ++v) acc2[v] = bb;
            }
            for (int kt = 0; kt < 8; ++kt) {
                v16h bf;
                if (PRE) {
                    bf = *reinterpret_cast<const v16h*>(
                        W2f + (((size_t)t * 2 + nt2) * 8 + kt) * 512 + lane * 16);
                } else {
                    #pragma unroll
                    for (int e = 0; e < 16; ++e) {
                        const int k = kt * 32 + e + 16 * khalf;
                        bf[e] = ok ? (_Float16)W2t[(size_t)k * OUT3 + n2] : (_Float16)0.0f;
                    }
                }
                const v16h af = *reinterpret_cast<const v16h*>(
                    &s_h1[(rt * 8 + kt) * 512 + lane * 16]);
                acc2 = __builtin_amdgcn_wmma_f32_16x16x32_f16(
                    false, af, false, bf, (short)0, acc2, false, false);
            }
            #pragma unroll
            for (int v = 0; v < 8; ++v) {
                const int rl = rt * 16 + v + 8 * khalf;
                s_p[rl * 32 + n2] = acc2[v];
            }
        }
        __syncthreads();

        // ---- Mixture phase: thread (r, j) per (row, component); width-8 shuffles ----
        {
            const int gr = rowBase + rM;
            const float rawa = s_p[rM * 32 + jM];
            const float rawb = s_p[rM * 32 + 8 + jM];
            const float raww = s_p[rM * 32 + 16 + jM];

            // softmax over j (width-8 butterfly)
            float mx = raww;
            mx = fmaxf(mx, __shfl_xor(mx, 1, 8));
            mx = fmaxf(mx, __shfl_xor(mx, 2, 8));
            mx = fmaxf(mx, __shfl_xor(mx, 4, 8));
            const float we = expf(raww - mx);
            float wsum = we;
            wsum += __shfl_xor(wsum, 1, 8);
            wsum += __shfl_xor(wsum, 2, 8);
            wsum += __shfl_xor(wsum, 4, 8);
            const float rho = 0.01f + 0.92f * we / wsum;

            const float thp  = s_thp[rM];
            const float phit = phi[(size_t)gr * LSEQ + (t + 1)];
            const float d    = phit - thp;
            const float x    = (d - TWO_PI_F * floorf(d * INV_TWO_PI_F)) * INV_TWO_PI_F;

            const float alpha = 2.0f * softplus_f(rawa);
            const float beta  = rawb - floorf(rawb);
            float u = x - beta; u -= floorf(u);
            u = fminf(fmaxf(u, EPS_F), 1.0f - EPS_F);
            const float lu = logf(u), l1 = log1pf(-u);
            const float A = alpha * lu, Bv = alpha * l1;
            const float hi = fmaxf(A, Bv), lo = fminf(A, Bv);
            const float lse = hi + log1pf(expf(lo - hi));
            const float S   = expf(A - lse);
            const float dS  = alpha * expf((alpha - 1.0f) * (lu + l1) - 2.0f * lse);

            float mterm = rho * (S - u);
            float dterm = rho * dS;
            mterm += __shfl_xor(mterm, 1, 8);
            mterm += __shfl_xor(mterm, 2, 8);
            mterm += __shfl_xor(mterm, 4, 8);
            dterm += __shfl_xor(dterm, 1, 8);
            dterm += __shfl_xor(dterm, 2, 8);
            dterm += __shfl_xor(dterm, 4, 8);

            if (jM == 0) {
                float y = x + mterm; y -= floorf(y);
                ldjAcc += logf(dterm);
                float tht = TWO_PI_F * y + thp;
                tht -= TWO_PI_F * floorf(tht * INV_TWO_PI_F);
                outTheta[(size_t)gr * LSEQ + (t + 1)] = tht;
                s_thp[rM] = tht;
            }
        }
        __syncthreads();
    }

    if (jM == 0) outLdj[rowBase + rM] = ldjAcc;
}

extern "C" void kernel_launch(void* const* d_in, const int* in_sizes, int n_in,
                              void* d_out, int out_size, void* d_ws, size_t ws_size,
                              hipStream_t stream) {
    (void)in_sizes; (void)n_in; (void)out_size;
    const float* phi = (const float*)d_in[0];
    const float* W0  = (const float*)d_in[1];
    const float* b0  = (const float*)d_in[2];
    const float* W1  = (const float*)d_in[3];
    const float* b1  = (const float*)d_in[4];
    const float* W2  = (const float*)d_in[5];
    const float* b2  = (const float*)d_in[6];
    float* outTheta = (float*)d_out;
    float* outLdj   = outTheta + (size_t)NB * LSEQ;

    dim3 grid(NB / ROWS);   // 128 blocks
    dim3 block(256);        // 8 wave32 waves

    if (d_ws != nullptr && ws_size >= WS_NEEDED) {
        _Float16* W1f = (_Float16*)d_ws;
        _Float16* W2f = W1f + W1F_HALVES;
        preswizzle_w1<<<dim3((TSTEPS * 16 * 8 * 32) / 256), block, 0, stream>>>(W1, W1f);
        preswizzle_w2<<<dim3((TSTEPS * 2 * 8 * 32) / 256), block, 0, stream>>>(W2, W2f);
        ar_flow_kernel<true><<<grid, block, 0, stream>>>(phi, W0, b0, W1, b1, W2, b2,
                                                         W1f, W2f, outTheta, outLdj);
    } else {
        ar_flow_kernel<false><<<grid, block, 0, stream>>>(phi, W0, b0, W1, b1, W2, b2,
                                                          nullptr, nullptr, outTheta, outLdj);
    }
}
